// W8A8BFP32O32LinearSiLU_4844723110246
// MI455X (gfx1250) — 8-bit matrix operand order repaired and checked by reading, compiled, not run on hardware
//
#include <hip/hip_runtime.h>
#include <hip/hip_bf16.h>
#include <cstdint>

typedef __attribute__((ext_vector_type(8))) int          v8i;
typedef __attribute__((ext_vector_type(4))) int          v4i;
typedef __attribute__((ext_vector_type(2))) int          v2i;
typedef __attribute__((ext_vector_type(4))) unsigned int v4u;

#define M_TOT 8192
#define N_TOT 11008
#define K_TOT 4096
#define BM 128
#define BN 128
#define BK 64
#define KSTEPS (K_TOT / BK)

#if defined(__has_builtin)
#if __has_builtin(__builtin_amdgcn_tensor_load_to_lds)
#define USE_TDM 1
#endif
#endif

#ifdef USE_TDM
__device__ __forceinline__ void tdm_load_tile(uint32_t lds_off, const void* g,
                                              uint32_t rows) {
    const uint64_t ga = (uint64_t)(uintptr_t)g;
    v4u g0 = { 1u,
               lds_off,
               (uint32_t)ga,
               ((uint32_t)(ga >> 32) & 0x01FFFFFFu) |
               (2u << 30) };
    v8i g1 = { 0,
               (int)((K_TOT & 0xFFFFu) << 16),
               (int)(((uint32_t)K_TOT >> 16) | ((rows & 0xFFFFu) << 16)),
               (int)((rows >> 16) | ((uint32_t)BK << 16)),
               (int)BM,
               (int)K_TOT,
               0, 0 };
    v4i z4 = {0, 0, 0, 0};
#if __clang_major__ >= 23
    v8i z8 = {0, 0, 0, 0, 0, 0, 0, 0};
    __builtin_amdgcn_tensor_load_to_lds(g0, g1, z4, z4, z8, 0);
#else
    __builtin_amdgcn_tensor_load_to_lds(g0, g1, z4, z4, 0);
#endif
}
#endif

__global__ __launch_bounds__(256)
void w8a8_silu_wmma(const int8_t* __restrict__ X,
                    const int8_t* __restrict__ W,
                    const float*  __restrict__ bias,
                    const float*  __restrict__ alpha_p,
                    float*        __restrict__ out)
{
    __shared__ __align__(16) int8_t As[2 * BM * BK];
    __shared__ __align__(16) int8_t Bs[2 * BN * BK];

    const int tid  = threadIdx.x;
    const int lane = tid & 31;
    const int wave = tid >> 5;
    const int wm   = wave >> 2;
    const int wn   = wave & 3;

    const int bm = blockIdx.y;
    const int bn = blockIdx.x;

    const int8_t* Xtile = X + (size_t)(bm * BM) * K_TOT;
    const int8_t* Wtile = W + (size_t)(bn * BN) * K_TOT;

    const v8i vzero = {0, 0, 0, 0, 0, 0, 0, 0};
    v8i acc[4][2];
#pragma unroll
    for (int i = 0; i < 4; ++i)
#pragma unroll
        for (int j = 0; j < 2; ++j) acc[i][j] = vzero;

#ifdef USE_TDM
    const uint32_t as_off = (uint32_t)(uintptr_t)(&As[0]);
    const uint32_t bs_off = (uint32_t)(uintptr_t)(&Bs[0]);

    if (wave == 0) {
        tdm_load_tile(as_off, Xtile, M_TOT - bm * BM);
        tdm_load_tile(bs_off, Wtile, N_TOT - bn * BN);
        __builtin_amdgcn_s_wait_tensorcnt(0);
    }
    __syncthreads();
#else
    const int lrow = tid >> 1;
    const int lkof = (tid & 1) * 32;
    const int8_t* ag = Xtile + (size_t)lrow * K_TOT + lkof;
    const int8_t* bg = Wtile + (size_t)lrow * K_TOT + lkof;
    {
        v4i a0 = *(const v4i*)(ag);
        v4i a1 = *(const v4i*)(ag + 16);
        v4i b0 = *(const v4i*)(bg);
        v4i b1 = *(const v4i*)(bg + 16);
        *(v4i*)(As + lrow * BK + lkof)      = a0;
        *(v4i*)(As + lrow * BK + lkof + 16) = a1;
        *(v4i*)(Bs + lrow * BK + lkof)      = b0;
        *(v4i*)(Bs + lrow * BK + lkof + 16) = b1;
    }
    __syncthreads();
#endif

    const int half      = lane >> 4;
    const int arow_base = wm * 64 + (lane & 15);
    const int akbase    = half * 16;
    const int bcol_base = wn * 32 + (lane & 15);
    const int bkhalf    = half * 16;

    for (int kk = 0; kk < KSTEPS; ++kk) {
        const int cur = kk & 1;
        const int nxt = cur ^ 1;

#ifdef USE_TDM
        if ((kk + 1 < KSTEPS) && wave == 0) {
            tdm_load_tile(as_off + (uint32_t)(nxt * BM * BK),
                          Xtile + (size_t)(kk + 1) * BK, M_TOT - bm * BM);
            tdm_load_tile(bs_off + (uint32_t)(nxt * BN * BK),
                          Wtile + (size_t)(kk + 1) * BK, N_TOT - bn * BN);
        }
#else
        v4i a0, a1, b0, b1;
        if (kk + 1 < KSTEPS) {
            const int8_t* agn = ag + (size_t)(kk + 1) * BK;
            const int8_t* bgn = bg + (size_t)(kk + 1) * BK;
            a0 = *(const v4i*)(agn);
            a1 = *(const v4i*)(agn + 16);
            b0 = *(const v4i*)(bgn);
            b1 = *(const v4i*)(bgn + 16);
        }
#endif

        const int8_t* abuf = As + cur * (BM * BK);
        const int8_t* bbuf = Bs + cur * (BN * BK);

        v8i afrag[4];
#pragma unroll
        for (int i = 0; i < 4; ++i) {
            const int8_t* ap = abuf + (arow_base + i * 16) * BK + akbase;
            v4i d0 = *(const v4i*)(ap);
            v4i d1 = *(const v4i*)(ap + 32);
            v8i t = {d0.x, d0.y, d0.z, d0.w, d1.x, d1.y, d1.z, d1.w};
            afrag[i] = t;
        }
        v8i bfrag[2];
#pragma unroll
        for (int j = 0; j < 2; ++j) {
            const int8_t* bp = bbuf + (bcol_base + j * 16) * BK + bkhalf;
            v4i lo = *(const v4i*)(bp);
            v4i hi = *(const v4i*)(bp + 32);
            v8i t = {lo.x, lo.y, lo.z, lo.w, hi.x, hi.y, hi.z, hi.w};
            bfrag[j] = t;
        }

#pragma unroll
        for (int i = 0; i < 4; ++i)
#pragma unroll
            for (int j = 0; j < 2; ++j)
                acc[i][j] = __builtin_amdgcn_wmma_i32_16x16x64_iu8(
                    true, afrag[i], true, bfrag[j], acc[i][j], false, false);

#ifdef USE_TDM
        if ((kk + 1 < KSTEPS) && wave == 0)
            __builtin_amdgcn_s_wait_tensorcnt(0);
#else
        if (kk + 1 < KSTEPS) {
            int8_t* asn = As + nxt * (BM * BK);
            int8_t* bsn = Bs + nxt * (BM * BK);
            *(v4i*)(asn + lrow * BK + lkof)      = a0;
            *(v4i*)(asn + lrow * BK + lkof + 16) = a1;
            *(v4i*)(bsn + lrow * BK + lkof)      = b0;
            *(v4i*)(bsn + lrow * BK + lkof + 16) = b1;
        }
#endif
        __syncthreads();
    }

    const float alpha    = *alpha_p;
    const int   row_lane = half * 8;
    const int   col_lane = lane & 15;
#pragma unroll
    for (int j = 0; j < 2; ++j) {
        const int   col = bn * BN + wn * 32 + j * 16 + col_lane;
        const float bv  = bias[col];
#pragma unroll
        for (int i = 0; i < 4; ++i) {
            const int row0 = bm * BM + wm * 64 + i * 16 + row_lane;
#pragma unroll
            for (int r = 0; r < 8; ++r) {
                float y = alpha * (float)acc[i][j][r] + bv;
                float s = y / (1.0f + __expf(-y));
                __builtin_nontemporal_store(s, out + (size_t)(row0 + r) * N_TOT + col);
            }
        }
    }
}

extern "C" void kernel_launch(void* const* d_in, const int* in_sizes, int n_in,
                              void* d_out, int out_size, void* d_ws, size_t ws_size,
                              hipStream_t stream) {
    const int8_t* X     = (const int8_t*)d_in[0];
    const int8_t* W     = (const int8_t*)d_in[1];
    const float*  bias  = (const float*)d_in[2];
    const float*  alpha = (const float*)d_in[3];
    float*        out   = (float*)d_out;

    dim3 grid(N_TOT / BN, M_TOT / BM);
    w8a8_silu_wmma<<<grid, 256, 0, stream>>>(X, W, bias, alpha, out);
}
